// LiquidOperator_80711025426845
// MI455X (gfx1250) — matrix operand order repaired and checked by reading, not compiled, not run on hardware
//
#include <hip/hip_runtime.h>
#include <math.h>

typedef __attribute__((ext_vector_type(16))) __bf16 v16bf;
typedef __attribute__((ext_vector_type(8)))  float  v8f;

#define WAVES 4
#define BLOCK 128
#define LDK 264   // bf16 LDS row stride (pad vs 256 to break bank conflicts)
#define LDO 260   // f32  LDS row stride

// ---------- bf16 helpers (RNE convert) ----------
__device__ __forceinline__ __bf16 f2bf(float f) {
  union { float f; unsigned u; } cv; cv.f = f;
  unsigned u = cv.u + 0x7FFFu + ((cv.u >> 16) & 1u);
  union { unsigned short s; __bf16 b; } cv2; cv2.s = (unsigned short)(u >> 16);
  return cv2.b;
}
__device__ __forceinline__ float bf2f(__bf16 b) {
  union { __bf16 b; unsigned short s; } cv; cv.b = b;
  union { unsigned u; float f; } cv2; cv2.u = ((unsigned)cv.s) << 16;
  return cv2.f;
}
__device__ __forceinline__ float silu_f(float x) { return x / (1.f + __expf(-x)); }

// ---------- WMMA fragment loads (contiguous k order, k = 16*(lane>>4) + i, wave32) ----------
// A: 16x32 bf16. lane m(0..15): K kk..kk+15; lane m+16: kk+16..kk+31.
__device__ __forceinline__ v16bf load_a(const __bf16* X, int kk, int lane) {
  int m = lane & 15, hi = lane >> 4;
  const __bf16* p0 = X + m * LDK + kk + 16 * hi;
  v16bf a;
#pragma unroll
  for (int i = 0; i < 16; ++i) a[i] = p0[i];
  return a;
}
// B: 32x16 bf16 from W[out][K] row-major (torch Linear). lane n: col=colbase+(lane&15),
// K segment kk+16*(lane>>4) .. +15 (contiguous in W).
template <int K>
__device__ __forceinline__ v16bf load_b(const __bf16* W, int colbase, int kk, int lane) {
  int n = lane & 15, hi = lane >> 4;
  const __bf16* p = W + (size_t)(colbase + n) * K + kk + 16 * hi;
  v16bf b;
#pragma unroll
  for (int i = 0; i < 16; ++i) b[i] = p[i];
  return b;
}

template <int K>
__device__ __forceinline__ v8f gemm_tile(const __bf16* Xlds, const __bf16* W, int colbase, int lane) {
  v8f acc = {0.f, 0.f, 0.f, 0.f, 0.f, 0.f, 0.f, 0.f};
#pragma unroll
  for (int kk = 0; kk < K; kk += 32) {
    v16bf a = load_a(Xlds, kk, lane);
    v16bf b = load_b<K>(W, colbase, kk, lane);
    acc = __builtin_amdgcn_wmma_f32_16x16x32_bf16(false, a, false, b, (short)0, acc, false, false);
  }
  return acc;
}

// D tile: VGPR j -> row j+8*(lane>>4), col colbase+(lane&15)
__device__ __forceinline__ void store_tile_lds(float* Y, int colbase, v8f acc,
                                               const float* bias, int lane) {
  int n = lane & 15, hi = lane >> 4;
  float bv = bias[colbase + n];
#pragma unroll
  for (int j = 0; j < 8; ++j) Y[(j + 8 * hi) * LDO + colbase + n] = acc[j] + bv;
}
__device__ __forceinline__ void store_tile_gbf(__bf16* G, int rowbase, int colbase, v8f acc,
                                               const float* bias, int lane) {
  int n = lane & 15, hi = lane >> 4;
  float bv = bias[colbase + n];
#pragma unroll
  for (int j = 0; j < 8; ++j)
    G[(size_t)(rowbase + j + 8 * hi) * 256 + colbase + n] = f2bf(acc[j] + bv);
}

template <int K>
__device__ __forceinline__ void block_gemm(const __bf16* Xlds, const __bf16* W,
                                           const float* bias, float* Ylds, int OUT) {
  int wave = threadIdx.x >> 5, lane = threadIdx.x & 31;
  for (int ct = wave; ct < OUT / 16; ct += WAVES) {
    v8f acc = gemm_tile<K>(Xlds, W, ct * 16, lane);
    store_tile_lds(Ylds, ct * 16, acc, bias, lane);
  }
}

// Row-wise LayerNorm of 16x256 f32 tile -> bf16 tile. 128 threads: 16 rows x 8 parts.
__device__ __forceinline__ void layernorm_to_bf16(const float* Yin, __bf16* Xout,
                                                  const float* g, const float* b,
                                                  float* redS, float* redQ) {
  int t = threadIdx.x, row = t >> 3, part = t & 7;
  const float* rp = Yin + row * LDO + part * 32;
  float s = 0.f, ss = 0.f;
#pragma unroll
  for (int i = 0; i < 32; ++i) { float x = rp[i]; s += x; ss += x * x; }
  redS[row * 8 + part] = s; redQ[row * 8 + part] = ss;
  __syncthreads();
  float tot = 0.f, totq = 0.f;
#pragma unroll
  for (int i = 0; i < 8; ++i) { tot += redS[row * 8 + i]; totq += redQ[row * 8 + i]; }
  float mean = tot * (1.f / 256.f);
  float var = totq * (1.f / 256.f) - mean * mean;
  float rs = rsqrtf(var + 1e-5f);
#pragma unroll
  for (int i = 0; i < 32; ++i) {
    int col = part * 32 + i;
    Xout[row * LDK + col] = f2bf((rp[i] - mean) * rs * g[col] + b[col]);
  }
}

// ---------- Kernel 1: f32 -> bf16 weight convert with K padding ----------
__global__ void convert_w_bf16(const float* __restrict__ src, __bf16* __restrict__ dst,
                               int rows, int K, int Kp) {
  int i = blockIdx.x * blockDim.x + threadIdx.x;
  if (i >= rows * Kp) return;
  int r = i / Kp, k = i - r * Kp;
  dst[i] = (k < K) ? f2bf(src[(size_t)r * K + k]) : f2bf(0.f);
}

// ---------- Kernel 2: per-timestep tokens -> K,V (bf16) ----------
__global__ void __launch_bounds__(BLOCK)
kv_precompute(const float* __restrict__ h_states, const __bf16* __restrict__ WbtpBF,
              const float* __restrict__ btp_b, const __bf16* __restrict__ WattnBF,
              const float* __restrict__ attn_in_b, __bf16* __restrict__ Kbf,
              __bf16* __restrict__ Vbf) {
  __shared__ __bf16 Xbf[16 * LDK];
  __shared__ float Yf[16 * LDO];
  int rowbase = blockIdx.x * 16;  // rows over T*S = 16384
  for (int e = threadIdx.x; e < 16 * 256; e += BLOCK) {
    int r = e >> 8, k = e & 255;
    Xbf[r * LDK + k] = f2bf(h_states[(size_t)(rowbase + r) * 256 + k]);
  }
  __syncthreads();
  block_gemm<256>(Xbf, WbtpBF, btp_b, Yf, 256);   // tokens (no activation)
  __syncthreads();
  for (int e = threadIdx.x; e < 16 * 256; e += BLOCK) {
    int r = e >> 8, k = e & 255;
    Xbf[r * LDK + k] = f2bf(Yf[r * LDO + k]);
  }
  __syncthreads();
  int wave = threadIdx.x >> 5, lane = threadIdx.x & 31;
  for (int ct = wave; ct < 16; ct += WAVES) {
    v8f aK = gemm_tile<256>(Xbf, WattnBF + 65536, ct * 16, lane);   // wk
    store_tile_gbf(Kbf, rowbase, ct * 16, aK, attn_in_b + 256, lane);
    v8f aV = gemm_tile<256>(Xbf, WattnBF + 131072, ct * 16, lane);  // wv
    store_tile_gbf(Vbf, rowbase, ct * 16, aV, attn_in_b + 512, lane);
  }
}

// ---------- Kernel 3: trunk MLP + q projection ----------
__global__ void __launch_bounds__(BLOCK)
trunk_kernel(const float* __restrict__ xy, const float* __restrict__ t_q,
             const int* __restrict__ comp, const float* __restrict__ sensor_time,
             const float* __restrict__ Bmat, const float* __restrict__ time_w,
             const float* __restrict__ time_b, const float* __restrict__ comp_emb,
             const __bf16* __restrict__ WtinBF, const float* __restrict__ trunk_in_b,
             const float* __restrict__ blk_ln_g, const float* __restrict__ blk_ln_b,
             const __bf16* __restrict__ WFc1BF, const float* __restrict__ blk_fc1_b,
             const __bf16* __restrict__ WFc2BF, const float* __restrict__ blk_fc2_b,
             const float* __restrict__ bn_g, const float* __restrict__ bn_b,
             const __bf16* __restrict__ WqBF, const float* __restrict__ attn_in_b,
             float* __restrict__ featg, float* __restrict__ qg, int* __restrict__ idxg) {
  __shared__ __bf16 Xbf[16 * LDK];
  __shared__ float Yf[16 * LDO];
  __shared__ float Ff[16 * LDO];
  __shared__ float redS[16 * 8], redQ[16 * 8];
  int rowbase = blockIdx.x * 16;
  {
    int row = threadIdx.x & 15, lanecol = threadIdx.x >> 4;  // 8 col-groups per row
    int n = rowbase + row;
    float x = xy[2 * n], y = xy[2 * n + 1], tq = t_q[n];
    int lo = 0, hi = 512;                                   // searchsorted 'right'
    while (lo < hi) { int mid = (lo + hi) >> 1; if (sensor_time[mid] <= tq) lo = mid + 1; else hi = mid; }
    int idx = lo - 1; idx = idx < 0 ? 0 : (idx > 511 ? 511 : idx);
    float dt = tq - sensor_time[idx]; if (dt < 0.f) dt = 0.f;
    if (lanecol == 0) idxg[n] = idx;
    int cc = comp[n];
    for (int col = lanecol; col < 160; col += 8) {
      float v;
      if (col < 64)      { float p = x * Bmat[col] + y * Bmat[64 + col]; v = cosf(p); }
      else if (col < 128){ int j = col - 64; float p = x * Bmat[j] + y * Bmat[64 + j]; v = sinf(p); }
      else if (col < 144){ int d = col - 128; v = dt * time_w[d] + time_b[d]; }
      else if (col < 152){ v = comp_emb[cc * 8 + (col - 144)]; }
      else               { v = 0.f; }                       // K pad 152->160
      Xbf[row * LDK + col] = f2bf(v);
    }
  }
  __syncthreads();
  block_gemm<160>(Xbf, WtinBF, trunk_in_b, Yf, 256);
  __syncthreads();
  for (int e = threadIdx.x; e < 16 * 256; e += BLOCK) {
    int r = e >> 8, k = e & 255;
    Ff[r * LDO + k] = silu_f(Yf[r * LDO + k]);
  }
  __syncthreads();
  for (int i = 0; i < 2; ++i) {
    layernorm_to_bf16(Ff, Xbf, blk_ln_g + i * 256, blk_ln_b + i * 256, redS, redQ);
    __syncthreads();
    block_gemm<256>(Xbf, WFc1BF + i * 65536, blk_fc1_b + i * 256, Yf, 256);
    __syncthreads();
    for (int e = threadIdx.x; e < 16 * 256; e += BLOCK) {
      int r = e >> 8, k = e & 255;
      Xbf[r * LDK + k] = f2bf(silu_f(Yf[r * LDO + k]));
    }
    __syncthreads();
    block_gemm<256>(Xbf, WFc2BF + i * 65536, blk_fc2_b + i * 256, Yf, 256);
    __syncthreads();
    for (int e = threadIdx.x; e < 16 * 256; e += BLOCK) {
      int r = e >> 8, k = e & 255;
      Ff[r * LDO + k] += Yf[r * LDO + k];
    }
    __syncthreads();
  }
  for (int e = threadIdx.x; e < 16 * 256; e += BLOCK) {
    int r = e >> 8, k = e & 255;
    featg[(size_t)(rowbase + r) * 256 + k] = Ff[r * LDO + k];
  }
  layernorm_to_bf16(Ff, Xbf, bn_g, bn_b, redS, redQ);   // q_in = LN(feat)
  __syncthreads();
  block_gemm<256>(Xbf, WqBF, attn_in_b, Yf, 256);       // q = q_in @ wq^T + bq
  __syncthreads();
  for (int e = threadIdx.x; e < 16 * 256; e += BLOCK) {
    int r = e >> 8, k = e & 255;
    qg[(size_t)(rowbase + r) * 256 + k] = Yf[r * LDO + k];
  }
}

// ---------- Kernel 4: single-query cross attention (one wave per query-head) ----------
__global__ void __launch_bounds__(BLOCK)
attn_kernel(const float* __restrict__ qg, const __bf16* __restrict__ Kbf,
            const __bf16* __restrict__ Vbf, const int* __restrict__ idxg,
            float* __restrict__ ctxg) {
  int gw = blockIdx.x * WAVES + (threadIdx.x >> 5);
  int lane = threadIdx.x & 31;
  int n = gw >> 2, h = gw & 3;
  int t = idxg[n];
  size_t kvbase = (size_t)t * 32 * 256 + h * 64;
  const float* qp = qg + (size_t)n * 256 + h * 64;
  const __bf16* kp = Kbf + kvbase + (size_t)lane * 256;   // lane = token s
  float s = 0.f;
#pragma unroll 8
  for (int d = 0; d < 64; ++d) s += qp[d] * bf2f(kp[d]);
  s *= 0.125f;                                            // 1/sqrt(64)
  float m = s;
  for (int off = 16; off > 0; off >>= 1) m = fmaxf(m, __shfl_xor(m, off, 32));
  float e = __expf(s - m);
  float sum = e;
  for (int off = 16; off > 0; off >>= 1) sum += __shfl_xor(sum, off, 32);
  float attn = e / sum;
  float acc0 = 0.f, acc1 = 0.f;                           // lane = dim d, d+32
  for (int si = 0; si < 32; ++si) {
    float a = __shfl(attn, si, 32);
    const __bf16* vp = Vbf + kvbase + (size_t)si * 256;
    acc0 += a * bf2f(vp[lane]);
    acc1 += a * bf2f(vp[lane + 32]);
  }
  ctxg[(size_t)n * 256 + h * 64 + lane] = acc0;
  ctxg[(size_t)n * 256 + h * 64 + lane + 32] = acc1;
}

// ---------- Kernel 5: attn_out + bc block + fused basis dot ----------
__global__ void __launch_bounds__(BLOCK)
out_kernel(const float* __restrict__ ctxg, const float* __restrict__ featg,
           const int* __restrict__ comp, const __bf16* __restrict__ WaoBF,
           const float* __restrict__ attn_out_b, const float* __restrict__ bc_ln_g,
           const float* __restrict__ bc_ln_b, const __bf16* __restrict__ Wbc1BF,
           const float* __restrict__ bc_b1, const __bf16* __restrict__ Wbc2BF,
           const float* __restrict__ bc_b2, const __bf16* __restrict__ WtoutBF,
           const float* __restrict__ trunk_out_b, const __bf16* __restrict__ WbprojBF,
           const float* __restrict__ bproj_b, const float* __restrict__ log_temp,
           const float* __restrict__ comp_scale, const float* __restrict__ comp_bias,
           float* __restrict__ out) {
  __shared__ __bf16 Xbf[16 * LDK];
  __shared__ __bf16 Fbf[16 * LDK];
  __shared__ float Yf[16 * LDO];
  __shared__ float Cf[16 * LDO];
  __shared__ float redS[16 * 8], redQ[16 * 8];
  __shared__ float dotv[16];
  int rowbase = blockIdx.x * 16;
  for (int e = threadIdx.x; e < 16 * 256; e += BLOCK) {
    int r = e >> 8, k = e & 255;
    Xbf[r * LDK + k] = f2bf(ctxg[(size_t)(rowbase + r) * 256 + k]);
    Fbf[r * LDK + k] = f2bf(featg[(size_t)(rowbase + r) * 256 + k]);
  }
  if (threadIdx.x < 16) dotv[threadIdx.x] = 0.f;
  __syncthreads();
  block_gemm<256>(Xbf, WaoBF, attn_out_b, Yf, 256);       // ctx @ attn_out^T + b
  __syncthreads();
  for (int e = threadIdx.x; e < 16 * 256; e += BLOCK) {
    int r = e >> 8, k = e & 255;
    Cf[r * LDO + k] = Yf[r * LDO + k];
  }
  __syncthreads();
  layernorm_to_bf16(Cf, Xbf, bc_ln_g, bc_ln_b, redS, redQ);
  __syncthreads();
  block_gemm<256>(Xbf, Wbc1BF, bc_b1, Yf, 256);
  __syncthreads();
  for (int e = threadIdx.x; e < 16 * 256; e += BLOCK) {
    int r = e >> 8, k = e & 255;
    Xbf[r * LDK + k] = f2bf(silu_f(Yf[r * LDO + k]));
  }
  __syncthreads();
  block_gemm<256>(Xbf, Wbc2BF, bc_b2, Yf, 256);
  __syncthreads();
  for (int e = threadIdx.x; e < 16 * 256; e += BLOCK) {
    int r = e >> 8, k = e & 255;
    float v = Cf[r * LDO + k] + Yf[r * LDO + k];          // residual
    Xbf[r * LDK + k] = f2bf(v);                           // branch input (bf16)
  }
  __syncthreads();
  {  // fused trunk/branch basis tiles + component-selected rank contraction
    int wave = threadIdx.x >> 5, lane = threadIdx.x & 31;
    int nIdx = lane & 15, hi = lane >> 4;
    int cr[8]; float rowsum[8];
#pragma unroll
    for (int j = 0; j < 8; ++j) { cr[j] = comp[rowbase + j + 8 * hi]; rowsum[j] = 0.f; }
    for (int ct = wave; ct < 48; ct += WAVES) {           // 768 cols = 3 comps x 256
      int colbase = ct * 16, compId = colbase >> 8;
      v8f at = gemm_tile<256>(Fbf, WtoutBF, colbase, lane);
      v8f ab = gemm_tile<256>(Xbf, WbprojBF, colbase, lane);
      float tb = trunk_out_b[colbase + nIdx];
      float bb = bproj_b[colbase + nIdx];
#pragma unroll
      for (int j = 0; j < 8; ++j)
        if (cr[j] == compId) rowsum[j] += (at[j] + tb) * (ab[j] + bb);
    }
#pragma unroll
    for (int j = 0; j < 8; ++j) atomicAdd(&dotv[j + 8 * hi], rowsum[j]);
  }
  __syncthreads();
  if (threadIdx.x < 16) {
    int n = rowbase + threadIdx.x, cc = comp[n];
    float temp = __expf(log_temp[0]);
    out[n] = dotv[threadIdx.x] * temp * comp_scale[cc] + comp_bias[cc];
  }
}

// ---------- workspace layout (WB_*: bf16 elements, OFF_*: bytes) ----------
#define WB_BTP   0
#define WB_ATTN  65536
#define WB_TIN   262144
#define WB_FC1   303104
#define WB_FC2   434176
#define WB_AOUT  565248
#define WB_BC1   630784
#define WB_BC2   696320
#define WB_TOUT  761856
#define WB_BPROJ 958464
#define OFF_WB   0u
#define OFF_K    2310144u
#define OFF_V    10698752u
#define OFF_FEAT 19087360u
#define OFF_Q    27475968u
#define OFF_CTX  35864576u
#define OFF_IDX  44253184u

extern "C" void kernel_launch(void* const* d_in, const int* in_sizes, int n_in,
                              void* d_out, int out_size, void* d_ws, size_t ws_size,
                              hipStream_t stream) {
  const float* xy          = (const float*)d_in[0];
  const float* t_q         = (const float*)d_in[1];
  const int*   comp        = (const int*)d_in[2];
  const float* h_states    = (const float*)d_in[3];
  const float* sensor_time = (const float*)d_in[4];
  const float* Bmat        = (const float*)d_in[5];
  const float* time_w      = (const float*)d_in[6];
  const float* time_b      = (const float*)d_in[7];
  const float* comp_emb    = (const float*)d_in[8];
  const float* trunk_in_w  = (const float*)d_in[9];
  const float* trunk_in_b  = (const float*)d_in[10];
  const float* blk_ln_g    = (const float*)d_in[11];
  const float* blk_ln_b    = (const float*)d_in[12];
  const float* blk_fc1_w   = (const float*)d_in[13];
  const float* blk_fc1_b   = (const float*)d_in[14];
  const float* blk_fc2_w   = (const float*)d_in[15];
  const float* blk_fc2_b   = (const float*)d_in[16];
  const float* btp_w       = (const float*)d_in[17];
  const float* btp_b       = (const float*)d_in[18];
  const float* bn_g        = (const float*)d_in[19];
  const float* bn_b        = (const float*)d_in[20];
  const float* attn_in_w   = (const float*)d_in[21];
  const float* attn_in_b   = (const float*)d_in[22];
  const float* attn_out_w  = (const float*)d_in[23];
  const float* attn_out_b  = (const float*)d_in[24];
  const float* bc_ln_g     = (const float*)d_in[25];
  const float* bc_ln_b     = (const float*)d_in[26];
  const float* bc_w1       = (const float*)d_in[27];
  const float* bc_b1       = (const float*)d_in[28];
  const float* bc_w2       = (const float*)d_in[29];
  const float* bc_b2       = (const float*)d_in[30];
  const float* trunk_out_w = (const float*)d_in[31];
  const float* trunk_out_b = (const float*)d_in[32];
  const float* bproj_w     = (const float*)d_in[33];
  const float* bproj_b     = (const float*)d_in[34];
  const float* log_temp    = (const float*)d_in[35];
  const float* comp_scale  = (const float*)d_in[36];
  const float* comp_bias   = (const float*)d_in[37];
  (void)in_sizes; (void)n_in; (void)ws_size;

  char* ws = (char*)d_ws;
  __bf16* WB  = (__bf16*)(ws + OFF_WB);
  __bf16* Kbf = (__bf16*)(ws + OFF_K);
  __bf16* Vbf = (__bf16*)(ws + OFF_V);
  float* featg = (float*)(ws + OFF_FEAT);
  float* qg    = (float*)(ws + OFF_Q);
  float* ctxg  = (float*)(ws + OFF_CTX);
  int*   idxg  = (int*)(ws + OFF_IDX);

  auto conv = [&](const float* src, __bf16* dst, int rows, int K, int Kp) {
    int total = rows * Kp;
    convert_w_bf16<<<(total + 255) / 256, 256, 0, stream>>>(src, dst, rows, K, Kp);
  };
  conv(btp_w,      WB + WB_BTP,   256, 256, 256);
  conv(attn_in_w,  WB + WB_ATTN,  768, 256, 256);
  conv(trunk_in_w, WB + WB_TIN,   256, 152, 160);
  conv(blk_fc1_w,  WB + WB_FC1,   512, 256, 256);
  conv(blk_fc2_w,  WB + WB_FC2,   512, 256, 256);
  conv(attn_out_w, WB + WB_AOUT,  256, 256, 256);
  conv(bc_w1,      WB + WB_BC1,   256, 256, 256);
  conv(bc_w2,      WB + WB_BC2,   256, 256, 256);
  conv(trunk_out_w,WB + WB_TOUT,  768, 256, 256);
  conv(bproj_w,    WB + WB_BPROJ, 768, 256, 256);

  kv_precompute<<<1024, BLOCK, 0, stream>>>(h_states, WB + WB_BTP, btp_b,
                                            WB + WB_ATTN, attn_in_b, Kbf, Vbf);
  trunk_kernel<<<512, BLOCK, 0, stream>>>(xy, t_q, comp, sensor_time, Bmat, time_w, time_b,
                                          comp_emb, WB + WB_TIN, trunk_in_b, blk_ln_g,
                                          blk_ln_b, WB + WB_FC1, blk_fc1_b, WB + WB_FC2,
                                          blk_fc2_b, bn_g, bn_b, WB + WB_ATTN, attn_in_b,
                                          featg, qg, idxg);
  attn_kernel<<<8192, BLOCK, 0, stream>>>(qg, Kbf, Vbf, idxg, ctxg);
  out_kernel<<<512, BLOCK, 0, stream>>>(ctxg, featg, comp, WB + WB_AOUT, attn_out_b,
                                        bc_ln_g, bc_ln_b, WB + WB_BC1, bc_b1, WB + WB_BC2,
                                        bc_b2, WB + WB_TOUT, trunk_out_b, WB + WB_BPROJ,
                                        bproj_b, log_temp, comp_scale, comp_bias,
                                        (float*)d_out);
}
